// TimeLSTM_MLP_3152505995670
// MI455X (gfx1250) — matrix operand order repaired and checked by reading, not compiled, not run on hardware
//
#include <hip/hip_runtime.h>
#include <hip/hip_bf16.h>
#include <math.h>

// Problem constants (from reference): B=1024, T=200, I=64, H=512, HID=256, NT=1
#define BB   1024
#define TT   200
#define II   64
#define HH   512
#define HIDN 256
#define KX   576    // H + I (fused gates GEMM K)
#define G4   2048   // 4*H

typedef __attribute__((ext_vector_type(16))) _Float16     v16h;
typedef __attribute__((ext_vector_type(8)))  float        v8f;
typedef __attribute__((ext_vector_type(4)))  unsigned int v4u;

// Explicit global-address-space pointer for B fragment loads so the backend
// emits global_load_b128 (LOADcnt only) instead of flat_load_b128.
typedef __attribute__((address_space(1))) const v4u gc_v4u;

union Frag32B { v4u u[2]; v16h v; };

// __builtin_amdgcn_global_load_async_to_lds_b128 takes
// (int4 addrspace(1)*, int4 addrspace(3)*, imm offset, imm cpol).
typedef int v4i __attribute__((vector_size(16)));
typedef __attribute__((address_space(1))) v4i global_v4i;
typedef __attribute__((address_space(3))) v4i lds_v4i;

__device__ __forceinline__ void async_copy_b128(const void* gp, const void* lp)
{
    // generic shared pointers carry the LDS byte offset in their low 32 bits
    __builtin_amdgcn_global_load_async_to_lds_b128(
        (global_v4i*)(uintptr_t)gp,
        (lds_v4i*)(uint32_t)(uintptr_t)lp,
        0, 0);
}

// ---------------------------------------------------------------------------
// Generic f16 WMMA GEMM: C[M,N] f32 = A[M,K] f16 (row-major, leading dim lda)
//                                    x Bsw (pre-swizzled f16 B fragments)
// Bsw layout: fragment for (ktile, ntile) = 32 lanes x 16 halves contiguous;
//   lane L holds column n = ntile*16 + (L&15), K rows ktile*32 + (L>>4)*16 + j
//   (contiguous k order; A fragments use the same order).
// Pipelining:
//   - A tile double-buffered in LDS via GLOBAL_LOAD_ASYNC_TO_LDS_B128
//     (ASYNCcnt; 2 async ops/wave/tile, in-order completion -> wait 2).
//   - All 8 B global_load_b128 of the tile are issued up front so the four
//     fragments' L2 round-trips overlap; WMMAs then run back-to-back with
//     staggered loadcnt waits.
// One 8-wave block per WGP (waves_per_eu(2)) so the allocator can keep
// 64 acc + 32 B-frag + 16 A-frag VGPRs live without scratch spills.
// grid = (N/128, M/128), block = 256 threads (8 waves).
// Each wave computes a 32(M) x 64(N) tile = 2x4 WMMA 16x16 accumulators.
// ---------------------------------------------------------------------------
__global__ void __launch_bounds__(256)
__attribute__((amdgpu_waves_per_eu(2)))
gemm_wmma_kernel(const _Float16* __restrict__ A, int lda,
                 const _Float16* __restrict__ Bsw, int ntilesN,
                 float* __restrict__ C, int ldc, int ktiles)
{
    __shared__ _Float16 smA[2][128 * 40];   // 2 x (128 rows x 32 halves, pad 40)

    const int tid  = threadIdx.x;
    const int lane = tid & 31;
    const int w    = tid >> 5;    // wave 0..7
    const int wr   = w & 3;       // M group (4 x 32 rows)
    const int wc   = w >> 2;      // N group (2 x 64 cols)
    const int m0   = blockIdx.y * 128;
    const int n0   = blockIdx.x * 128;
    const int g    = lane >> 4;   // lane half-group
    const int nl   = lane & 15;

    // Per-thread async-copy slots: 2 x 16B per tile (512 slots / 256 threads)
    const int row0 = tid >> 2;            // slot it=0
    const int ch0  = tid & 3;
    const int row1 = (tid + 256) >> 2;    // slot it=1
    const int ch1  = tid & 3;
    const _Float16* ga0 = A + (size_t)(m0 + row0) * lda + ch0 * 8;
    const _Float16* ga1 = A + (size_t)(m0 + row1) * lda + ch1 * 8;
    const int ldsOff0 = row0 * 40 + ch0 * 8;
    const int ldsOff1 = row1 * 40 + ch1 * 8;

    // Per-lane B fragment pointers (contiguous 32B per lane, L2-resident)
    gc_v4u* bp[4];
    #pragma unroll
    for (int nt = 0; nt < 4; ++nt) {
        int ntG = (n0 >> 4) + wc * 4 + nt;
        bp[nt] = (gc_v4u*)(uintptr_t)
                 ((const v4u*)Bsw + ((size_t)ntG * 32 + lane) * 2);
    }
    const size_t bstep = (size_t)ntilesN * 32 * 2;   // v4u per k-tile

    v8f acc[2][4] = {};

    // Prologue: async-stage A tile 0 into LDS buffer 0
    async_copy_b128(ga0, &smA[0][ldsOff0]);
    async_copy_b128(ga1, &smA[0][ldsOff1]);

    for (int kt = 0; kt < ktiles; ++kt) {
        const int buf = kt & 1;
        if (kt + 1 < ktiles) {
            // stage A tile kt+1 into the other LDS buffer
            async_copy_b128(ga0 + (kt + 1) * 32, &smA[buf ^ 1][ldsOff0]);
            async_copy_b128(ga1 + (kt + 1) * 32, &smA[buf ^ 1][ldsOff1]);
            __builtin_amdgcn_s_wait_asynccnt(2);   // A tile kt retired
        } else {
            __builtin_amdgcn_s_wait_asynccnt(0);
        }
        __syncthreads();

        // Issue all 8 B fragment loads of tile kt up front (overlapping
        // round-trips); waits before the WMMAs become staggered.
        Frag32B bcur[4];
        #pragma unroll
        for (int nt = 0; nt < 4; ++nt) {
            bcur[nt].u[0] = bp[nt][0];
            bcur[nt].u[1] = bp[nt][1];
            bp[nt] += bstep;
        }

        // A fragments: lane half g holds contiguous k = 16*g + 0..15
        Frag32B afr[2];
        #pragma unroll
        for (int mt = 0; mt < 2; ++mt) {
            int mr = wr * 32 + mt * 16 + nl;
            afr[mt].u[0] = *(const v4u*)(&smA[buf][mr * 40 + g * 16]);
            afr[mt].u[1] = *(const v4u*)(&smA[buf][mr * 40 + g * 16 + 8]);
        }

        #pragma unroll
        for (int nt = 0; nt < 4; ++nt)
            #pragma unroll
            for (int mt = 0; mt < 2; ++mt)
                acc[mt][nt] = __builtin_amdgcn_wmma_f32_16x16x32_f16(
                    false, afr[mt].v, false, bcur[nt].v,
                    (short)0, acc[mt][nt], false, false);

        __syncthreads();   // tile kt fully consumed before kt+2 overwrites buf
    }

    // Epilogue: C/D 16x16 f32 layout -> (row = 8*g + r, col = nl)
    #pragma unroll
    for (int mt = 0; mt < 2; ++mt) {
        #pragma unroll
        for (int nt = 0; nt < 4; ++nt) {
            int col = n0 + wc * 64 + nt * 16 + nl;
            int rw0 = m0 + wr * 32 + mt * 16 + g * 8;
            #pragma unroll
            for (int r = 0; r < 8; ++r)
                C[(size_t)(rw0 + r) * ldc + col] = acc[mt][nt][r];
        }
    }
}

// ---------------------------------------------------------------------------
// Weight pre-swizzle into WMMA B-fragment order
// ---------------------------------------------------------------------------
// Fused gates weight: B[k,n] = (k<512) ? W_all[n,k] : U_all[n,k-512]
__global__ void prep_Wg(const float* __restrict__ Wall,
                        const float* __restrict__ Uall,
                        _Float16* __restrict__ out)
{
    int idx = blockIdx.x * 256 + threadIdx.x;        // 18*128*32*16 elements
    if (idx >= 18 * 128 * 32 * 16) return;
    int j    = idx & 15;
    int lane = (idx >> 4) & 31;
    int nt   = (idx >> 9) & 127;
    int kt   = idx >> 16;
    int n = nt * 16 + (lane & 15);
    int k = kt * 32 + (lane >> 4) * 16 + j;
    float v = (k < HH) ? Wall[(size_t)n * HH + k] : Uall[(size_t)n * II + (k - HH)];
    out[idx] = (_Float16)v;
}

// Generic: B[k,n] = W[n,k], W row-major [N, Kdim]
__global__ void prep_Wsw(const float* __restrict__ W, int Kdim, int ntilesN,
                         int total, _Float16* __restrict__ out)
{
    int idx = blockIdx.x * 256 + threadIdx.x;
    if (idx >= total) return;
    int j    = idx & 15;
    int lane = (idx >> 4) & 31;
    int t    = idx >> 9;
    int nt   = t % ntilesN;
    int kt   = t / ntilesN;
    int n = nt * 16 + (lane & 15);
    int k = kt * 32 + (lane >> 4) * 16 + j;
    out[idx] = (_Float16)W[(size_t)n * Kdim + k];
}

// x [B,T,I] f32 -> xT [T,B,I] f16
__global__ void prep_xT(const float* __restrict__ x, _Float16* __restrict__ xT)
{
    int idx = blockIdx.x * 256 + threadIdx.x;
    if (idx >= BB * TT * II) return;
    int i = idx % II;
    int t = (idx / II) % TT;
    int b = idx / (II * TT);
    xT[((size_t)t * BB + b) * II + i] = (_Float16)x[idx];
}

// init hx = [h=0 | x_0], c16 = 0, c32 = 0
__global__ void prep_state(const _Float16* __restrict__ xT,
                           _Float16* __restrict__ hx,
                           _Float16* __restrict__ c16,
                           float* __restrict__ c32)
{
    int idx = blockIdx.x * 256 + threadIdx.x;        // B * 576
    if (idx >= BB * KX) return;
    int k = idx % KX;
    int b = idx / KX;
    if (k < HH) {
        hx[idx] = (_Float16)0.f;
        c16[(size_t)b * HH + k] = (_Float16)0.f;
        c32[(size_t)b * HH + k] = 0.f;
    } else {
        hx[idx] = xT[(size_t)b * II + (k - HH)];     // t = 0 slice
    }
}

__global__ void prep_bg(const float* __restrict__ ba, const float* __restrict__ bu,
                        float* __restrict__ bg)
{
    int i = blockIdx.x * 256 + threadIdx.x;
    if (i < G4) bg[i] = ba[i] + bu[i];
}

// ---------------------------------------------------------------------------
// Per-step cell update (sigmoid gates, time-discounted cell), stages x_{t+1}
// ---------------------------------------------------------------------------
__device__ __forceinline__ float sigmoidf_(float x) { return 1.f / (1.f + __expf(-x)); }

__global__ void step_pointwise(const float* __restrict__ graw,
                               const float* __restrict__ draw,
                               const float* __restrict__ bg,
                               const float* __restrict__ bd,
                               const float* __restrict__ ts,
                               const _Float16* __restrict__ xT,
                               float* __restrict__ c32,
                               _Float16* __restrict__ c16,
                               _Float16* __restrict__ hx,
                               int t)
{
    int idx = blockIdx.x * 256 + threadIdx.x;        // B * H
    if (idx >= BB * HH) return;
    int n = idx & (HH - 1);
    int b = idx >> 9;

    float c   = c32[idx];
    float cs1 = tanhf(draw[idx] + bd[n]);
    float tt  = ts[(size_t)b * TT + t];
    float cadj = c - cs1 + cs1 * tt;

    const float* gr = graw + (size_t)b * G4;
    float f  = sigmoidf_(gr[n]            + bg[n]);
    float ig = sigmoidf_(gr[HH + n]       + bg[HH + n]);
    float o  = sigmoidf_(gr[2 * HH + n]   + bg[2 * HH + n]);
    float ct = sigmoidf_(gr[3 * HH + n]   + bg[3 * HH + n]);

    float cn = f * cadj + ig * ct;
    float hn = o * tanhf(cn);

    c32[idx] = cn;
    c16[idx] = (_Float16)cn;
    hx[(size_t)b * KX + n] = (_Float16)hn;
    if (n < II && (t + 1) < TT)
        hx[(size_t)b * KX + HH + n] = xT[((size_t)(t + 1) * BB + b) * II + n];
}

// MLP tail: out[b] = b2 + sum_n relu(hid[b,n] + b1[n]) * W2[n]
__global__ void head2(const float* __restrict__ hid, const float* __restrict__ b1,
                      const float* __restrict__ W2, const float* __restrict__ b2,
                      float* __restrict__ out)
{
    int b = blockIdx.x * 256 + threadIdx.x;
    if (b >= BB) return;
    float s = b2[0];
    const float* hr = hid + (size_t)b * HIDN;
    for (int n = 0; n < HIDN; ++n) {
        float h = hr[n] + b1[n];
        s += (h > 0.f ? h : 0.f) * W2[n];
    }
    out[b] = s;
}

// ---------------------------------------------------------------------------
extern "C" void kernel_launch(void* const* d_in, const int* in_sizes, int n_in,
                              void* d_out, int out_size, void* d_ws, size_t ws_size,
                              hipStream_t stream)
{
    const float* x    = (const float*)d_in[0];
    const float* ts   = (const float*)d_in[1];
    const float* Wall = (const float*)d_in[2];
    const float* ball = (const float*)d_in[3];
    const float* Uall = (const float*)d_in[4];
    const float* bu   = (const float*)d_in[5];
    const float* Wd   = (const float*)d_in[6];
    const float* bd   = (const float*)d_in[7];
    const float* W1   = (const float*)d_in[8];
    const float* b1   = (const float*)d_in[9];
    const float* W2   = (const float*)d_in[10];
    const float* b2   = (const float*)d_in[11];

    char* ws = (char*)d_ws;
    size_t off = 0;
    auto alloc = [&](size_t bytes) -> void* {
        void* p = ws + off;
        off += (bytes + 255) & ~(size_t)255;
        return p;
    };

    _Float16* Wg_sw = (_Float16*)alloc((size_t)18 * 128 * 32 * 16 * 2);  // 2.36 MB
    _Float16* Wd_sw = (_Float16*)alloc((size_t)16 * 32 * 32 * 16 * 2);   // 0.5 MB
    _Float16* W1_sw = (_Float16*)alloc((size_t)16 * 16 * 32 * 16 * 2);   // 0.25 MB
    _Float16* xT    = (_Float16*)alloc((size_t)BB * TT * II * 2);        // 26 MB
    _Float16* hx    = (_Float16*)alloc((size_t)BB * KX * 2);             // 1.1 MB
    _Float16* c16   = (_Float16*)alloc((size_t)BB * HH * 2);             // 1 MB
    float*    c32   = (float*)   alloc((size_t)BB * HH * 4);             // 2 MB
    float*    bg    = (float*)   alloc((size_t)G4 * 4);
    float*    graw  = (float*)   alloc((size_t)BB * G4 * 4);             // 8 MB
    float*    draw  = (float*)   alloc((size_t)BB * HH * 4);             // 2 MB
    float*    hid   = (float*)   alloc((size_t)BB * HIDN * 4);           // 1 MB
    (void)ws_size;

    // --- one-time prep ---
    prep_Wg   <<<(18 * 128 * 32 * 16 + 255) / 256, 256, 0, stream>>>(Wall, Uall, Wg_sw);
    prep_Wsw  <<<(16 * 32 * 32 * 16 + 255) / 256, 256, 0, stream>>>(Wd, HH, 32,
                                                                    16 * 32 * 32 * 16, Wd_sw);
    prep_Wsw  <<<(16 * 16 * 32 * 16 + 255) / 256, 256, 0, stream>>>(W1, HH, 16,
                                                                    16 * 16 * 32 * 16, W1_sw);
    prep_xT   <<<(BB * TT * II + 255) / 256, 256, 0, stream>>>(x, xT);
    prep_state<<<(BB * KX + 255) / 256, 256, 0, stream>>>(xT, hx, c16, c32);
    prep_bg   <<<(G4 + 255) / 256, 256, 0, stream>>>(ball, bu, bg);

    // --- recurrence: 200 sequential steps ---
    for (int t = 0; t < TT; ++t) {
        // gates_raw[B, 4H] = [h | x_t] (K=576) x Wg
        gemm_wmma_kernel<<<dim3(G4 / 128, BB / 128), 256, 0, stream>>>(
            hx, KX, Wg_sw, G4 / 16, graw, G4, KX / 32);
        // d_raw[B, H] = c (K=512) x Wd
        gemm_wmma_kernel<<<dim3(HH / 128, BB / 128), 256, 0, stream>>>(
            c16, HH, Wd_sw, HH / 16, draw, HH, HH / 32);
        // cell update + stage x_{t+1}
        step_pointwise<<<(BB * HH + 255) / 256, 256, 0, stream>>>(
            graw, draw, bg, bd, ts, xT, c32, c16, hx, t);
    }

    // --- MLP head ---
    gemm_wmma_kernel<<<dim3(HIDN / 128, BB / 128), 256, 0, stream>>>(
        hx, KX, W1_sw, HIDN / 16, hid, HIDN, HH / 32);
    head2<<<(BB + 255) / 256, 256, 0, stream>>>(hid, b1, W2, b2, (float*)d_out);
}
